// RNNAutoEncoder_71588514890544
// MI455X (gfx1250) — compile-verified, matrix operand order checked by reading, not run on hardware
//
#include <hip/hip_runtime.h>
#include <hip/hip_bf16.h>
#include <math.h>
#include <stdint.h>

#define B_  256
#define T_  64
#define E_  1024
#define H_  1024
#define Z_  512
#define GH  (4 * H_)   // 4096

typedef __attribute__((ext_vector_type(4)))  float           f32x4;
typedef __attribute__((ext_vector_type(8)))  float           v8f;
typedef __attribute__((ext_vector_type(4)))  unsigned short  u16x4;
typedef __attribute__((ext_vector_type(8)))  unsigned short  u16x8;
typedef __attribute__((ext_vector_type(16))) __bf16          v16bf;

// round-to-nearest-even fp32 -> bf16 (bit pattern)
__device__ __forceinline__ unsigned short f2bf(float f) {
    unsigned u = __float_as_uint(f);
    return (unsigned short)((u + 0x7FFFu + ((u >> 16) & 1u)) >> 16);
}

// branch-free tanh / sigmoid via hardware exp (v_exp_f32)
__device__ __forceinline__ float fast_tanh(float x) {
    float e = __expf(2.0f * x);
    return 1.0f - 2.0f / (e + 1.0f);
}
__device__ __forceinline__ float fast_sigmoid(float x) {
    return 1.0f / (1.0f + __expf(-x));
}

__device__ __forceinline__ v16bf make_frag(u16x8 lo, u16x8 hi) {
    union { u16x8 u[2]; v16bf v; } c;
    c.u[0] = lo; c.u[1] = hi;
    return c.v;
}

// Generic shared pointer -> LDS byte address (aperture truncation: low 32 bits of the address)
__device__ __forceinline__ unsigned lds_addr(const void* p) {
    return (unsigned)(uintptr_t)p;
}

// Direct memory -> LDS async copy, 16B per lane, GVS addressing (SGPR base + VGPR byte offset).
// Tracked by ASYNCcnt.
__device__ __forceinline__ void async_g2l_b128(unsigned lds_dst, unsigned byte_off,
                                               const void* sbase) {
    asm volatile("global_load_async_to_lds_b128 %0, %1, %2 offset:0"
                 :: "v"(lds_dst), "v"(byte_off), "s"(sbase) : "memory");
}

__device__ __forceinline__ void wait_asynccnt0() {
    asm volatile("s_wait_asynccnt 0" ::: "memory");
}

// Out[M,N] = act( A[M,K](bf16) * W[N,K](bf16)^T + Cin + bias1 + bias2 )
// BM x 128 output tile per 256-thread block (8 wave32s), double-buffered async LDS staging.
// BM=128: wave = 32(M) x 64(N) -> 2x4 accums.  BM=64: wave = 32(M) x 32(N) -> 2x2 accums.
template<int BM>
__global__ __launch_bounds__(256)
void wmma_gemm_async(const unsigned short* __restrict__ A, int lda,
                     const unsigned short* __restrict__ W, int ldw,
                     const float* __restrict__ Cin, int ldc,
                     const float* __restrict__ bias1,
                     const float* __restrict__ bias2,
                     float* __restrict__ Out, int ldo,
                     int K, int act_tanh)
{
    constexpr int WNSH = (BM == 128) ? 1 : 2;    // bits of wave-id used for N placement
    constexpr int NJ   = (BM == 128) ? 4 : 2;    // 16-wide N tiles per wave
    constexpr unsigned A_BUF = (unsigned)BM * 40u * 2u;      // bytes per A buffer
    constexpr unsigned W_BUF = 128u * 40u * 2u;              // bytes per W buffer

    __shared__ unsigned short At[2][BM][40];     // 32 bf16 + 8 pad -> 80B rows (16B aligned)
    __shared__ unsigned short Wt[2][128][40];

    const int tid  = threadIdx.x;
    const int lane = tid & 31;
    const int wid  = tid >> 5;
    const int wn   = wid & ((1 << WNSH) - 1);
    const int wm   = wid >> WNSH;
    const int lrow = lane & 15;
    const int lkh  = lane >> 4;                  // 0/1: which K/M half this lane owns

    const int mbase = blockIdx.y * BM;
    const int nbase = blockIdx.x * 128;

    // ---- accumulator init from Cin / biases (C/D layout: M = r + 8*lkh, N = lane&15)
    v8f acc[2][NJ];
    #pragma unroll
    for (int i = 0; i < 2; ++i) {
        #pragma unroll
        for (int j = 0; j < NJ; ++j) {
            const int gm = mbase + wm * 32 + i * 16 + lkh * 8;
            const int gn = nbase + wn * (16 * NJ) + j * 16 + lrow;
            #pragma unroll
            for (int r = 0; r < 8; ++r) {
                float cv = 0.0f;
                if (Cin)   cv  = Cin[(long long)(gm + r) * ldc + gn];
                if (bias1) cv += bias1[gn];
                if (bias2) cv += bias2[gn];
                acc[i][j][r] = cv;
            }
        }
    }

    // uniform SGPR bases for the async copies
    const unsigned short* Abase = A + (size_t)mbase * lda;
    const unsigned short* Wbase = W + (size_t)nbase * ldw;
    const unsigned a_lds0 = lds_addr(&At[0][0][0]);
    const unsigned w_lds0 = lds_addr(&Wt[0][0][0]);

    // async-issue one K-tile (32 wide) into LDS buffer `buf`
    auto issue = [&](int k0, int buf) {
        const unsigned ab = a_lds0 + (unsigned)buf * A_BUF;
        const unsigned wb = w_lds0 + (unsigned)buf * W_BUF;
        #pragma unroll
        for (int it = 0; it < BM / 64; ++it) {       // A tile: BM*4 16B chunks
            const int idx = tid + it * 256;
            const int r   = idx >> 2;
            const int ce  = (idx & 3) * 8;           // element offset in row
            async_g2l_b128(ab + (unsigned)(r * 80 + ce * 2),
                           (unsigned)((r * lda + k0 + ce) * 2), Abase);
        }
        #pragma unroll
        for (int it = 0; it < 2; ++it) {             // W tile: 512 16B chunks
            const int idx = tid + it * 256;
            const int r   = idx >> 2;
            const int ce  = (idx & 3) * 8;
            async_g2l_b128(wb + (unsigned)(r * 80 + ce * 2),
                           (unsigned)((r * ldw + k0 + ce) * 2), Wbase);
        }
    };

    issue(0, 0);                                     // prologue: fill buffer 0

    for (int k0 = 0; k0 < K; k0 += 32) {
        const int cur = (k0 >> 5) & 1;

        wait_asynccnt0();          // this wave's copies into buf `cur` are done
        __syncthreads();           // => all waves' copies done; all reads of buf cur^1 retired

        if (k0 + 32 < K) issue(k0 + 32, cur ^ 1);    // overlap next copy with this compute

        // ---- fragment loads (16-bit A/B layout: lane<16 -> K{0..7,16..23}, lane>=16 -> K{8..15,24..31})
        v16bf afr[2], bfr[NJ];
        #pragma unroll
        for (int i = 0; i < 2; ++i) {
            const int rr = wm * 32 + i * 16 + lrow;
            u16x8 lo = *(const u16x8*)&At[cur][rr][lkh * 8];
            u16x8 hi = *(const u16x8*)&At[cur][rr][16 + lkh * 8];
            afr[i] = make_frag(lo, hi);
        }
        #pragma unroll
        for (int j = 0; j < NJ; ++j) {
            const int rr = wn * (16 * NJ) + j * 16 + lrow;
            u16x8 lo = *(const u16x8*)&Wt[cur][rr][lkh * 8];
            u16x8 hi = *(const u16x8*)&Wt[cur][rr][16 + lkh * 8];
            bfr[j] = make_frag(lo, hi);
        }

        #pragma unroll
        for (int i = 0; i < 2; ++i)
            #pragma unroll
            for (int j = 0; j < NJ; ++j)
                acc[i][j] = __builtin_amdgcn_wmma_f32_16x16x32_bf16(
                    false, afr[i], false, bfr[j], (short)0, acc[i][j], false, false);
    }

    // ---- epilogue
    #pragma unroll
    for (int i = 0; i < 2; ++i) {
        #pragma unroll
        for (int j = 0; j < NJ; ++j) {
            const int gm = mbase + wm * 32 + i * 16 + lkh * 8;
            const int gn = nbase + wn * (16 * NJ) + j * 16 + lrow;
            #pragma unroll
            for (int r = 0; r < 8; ++r) {
                float v = acc[i][j][r];
                if (act_tanh) v = fast_tanh(v);
                Out[(long long)(gm + r) * ldo + gn] = v;
            }
        }
    }
}

// one-time fp32 -> bf16 conversion, 4 elements / thread
__global__ void f32_to_bf16_v4(const float* __restrict__ in,
                               unsigned short* __restrict__ out, int n4)
{
    const int i = blockIdx.x * blockDim.x + threadIdx.x;
    if (i >= n4) return;
    f32x4 v = *(const f32x4*)(in + (size_t)i * 4);
    u16x4 o;
    o[0] = f2bf(v[0]); o[1] = f2bf(v[1]); o[2] = f2bf(v[2]); o[3] = f2bf(v[3]);
    *(u16x4*)(out + (size_t)i * 4) = o;
}

// pointwise LSTM cell update (gate order i,f,g,o); h written as bf16 for the next GEMM
__global__ void lstm_update(const float* __restrict__ gates,
                            unsigned short* __restrict__ h, float* __restrict__ c)
{
    const int idx = blockIdx.x * blockDim.x + threadIdx.x;
    if (idx >= B_ * H_) return;
    const int b = idx >> 10;           // H_ = 1024
    const int n = idx & (H_ - 1);
    const float* g = gates + (size_t)b * GH;
    const float ig = fast_sigmoid(g[n]);
    const float fg = fast_sigmoid(g[H_ + n]);
    const float gg = fast_tanh(g[2 * H_ + n]);
    const float og = fast_sigmoid(g[3 * H_ + n]);
    const float c2 = fg * c[idx] + ig * gg;
    c[idx] = c2;
    h[idx] = f2bf(og * fast_tanh(c2));
}

__global__ void init_state(unsigned short* __restrict__ h, float* __restrict__ c,
                           unsigned short* __restrict__ inp01)
{
    const int idx = blockIdx.x * blockDim.x + threadIdx.x;
    if (idx < B_ * H_) {
        h[idx] = 0;                 // bf16 +0.0
        c[idx] = 0.0f;
        inp01[idx] = f2bf(0.1f);
    }
}

extern "C" void kernel_launch(void* const* d_in, const int* in_sizes, int n_in,
                              void* d_out, int out_size, void* d_ws, size_t ws_size,
                              hipStream_t stream)
{
    (void)in_sizes; (void)n_in; (void)out_size; (void)ws_size;

    const float* x      = (const float*)d_in[0];   // [B,T,E]
    const float* W_ih_e = (const float*)d_in[1];   // [4H,E]
    const float* W_hh_e = (const float*)d_in[2];   // [4H,H]
    const float* b_ih_e = (const float*)d_in[3];
    const float* b_hh_e = (const float*)d_in[4];
    const float* W_ih_d = (const float*)d_in[5];   // [4H,H]
    const float* W_hh_d = (const float*)d_in[6];   // [4H,H]
    const float* b_ih_d = (const float*)d_in[7];
    const float* b_hh_d = (const float*)d_in[8];
    const float* Wz     = (const float*)d_in[9];   // [Z,H]
    const float* bz     = (const float*)d_in[10];
    float* out = (float*)d_out;                    // [B,T,Z] fp32

    // ---- workspace layout
    char* ws = (char*)d_ws;
    float*          Xp    = (float*)ws;                               // [B*T, 4H] fp32
    ws += (size_t)B_ * T_ * GH * sizeof(float);
    float*          gates = (float*)ws;   ws += (size_t)B_ * GH * sizeof(float);
    float*          c     = (float*)ws;   ws += (size_t)B_ * H_ * sizeof(float);
    unsigned short* hb    = (unsigned short*)ws; ws += (size_t)B_ * H_ * 2;
    unsigned short* inp01 = (unsigned short*)ws; ws += (size_t)B_ * H_ * 2;
    unsigned short* xb    = (unsigned short*)ws; ws += (size_t)B_ * T_ * E_ * 2;
    unsigned short* Wb_ih_e = (unsigned short*)ws; ws += (size_t)GH * E_ * 2;
    unsigned short* Wb_hh_e = (unsigned short*)ws; ws += (size_t)GH * H_ * 2;
    unsigned short* Wb_ih_d = (unsigned short*)ws; ws += (size_t)GH * H_ * 2;
    unsigned short* Wb_hh_d = (unsigned short*)ws; ws += (size_t)GH * H_ * 2;
    unsigned short* Wzb     = (unsigned short*)ws; ws += (size_t)Z_ * H_ * 2;

    const dim3 blk(256);
    const int  pw_grid = (B_ * H_ + 255) / 256;
    auto cvt = [&](const float* src, unsigned short* dst, int n) {
        f32_to_bf16_v4<<<(n / 4 + 255) / 256, blk, 0, stream>>>(src, dst, n / 4);
    };

    // ---- one-time bf16 conversions
    cvt(x,      xb,      B_ * T_ * E_);
    cvt(W_ih_e, Wb_ih_e, GH * E_);
    cvt(W_hh_e, Wb_hh_e, GH * H_);
    cvt(W_ih_d, Wb_ih_d, GH * H_);
    cvt(W_hh_d, Wb_hh_d, GH * H_);
    cvt(Wz,     Wzb,     Z_ * H_);
    init_state<<<pw_grid, blk, 0, stream>>>(hb, c, inp01);

    // ---- Xp = x @ W_ih_e^T + b_ih_e + b_hh_e  (one big parallel GEMM, M = B*T)
    {
        dim3 g(GH / 128, (B_ * T_) / 128);   // (32, 128)
        wmma_gemm_async<128><<<g, blk, 0, stream>>>(xb, E_, Wb_ih_e, E_, nullptr, 0,
                                                    b_ih_e, b_hh_e, Xp, GH, E_, 0);
    }

    const dim3 gg(GH / 128, B_ / 64);   // (32, 4) = 128 blocks for recurrent GEMMs
    const dim3 gz(Z_ / 128, B_ / 64);   // (4, 4)

    // ---- encoder scan
    for (int t = 0; t < T_; ++t) {
        // gates = h @ W_hh_e^T + Xp[:, t, :]   (Cin row b -> Xp[(b*T+t)*4H])
        wmma_gemm_async<64><<<gg, blk, 0, stream>>>(hb, H_, Wb_hh_e, H_,
                                                    Xp + (size_t)t * GH, T_ * GH,
                                                    nullptr, nullptr, gates, GH, H_, 0);
        lstm_update<<<pw_grid, blk, 0, stream>>>(gates, hb, c);
    }

    // ---- decoder scan
    for (int t = 0; t < T_; ++t) {
        const unsigned short* inp = (t == 0) ? inp01 : hb;
        // gates = inp @ W_ih_d^T + b_ih_d + b_hh_d
        wmma_gemm_async<64><<<gg, blk, 0, stream>>>(inp, H_, Wb_ih_d, H_, nullptr, 0,
                                                    b_ih_d, b_hh_d, gates, GH, H_, 0);
        // gates += h @ W_hh_d^T
        wmma_gemm_async<64><<<gg, blk, 0, stream>>>(hb, H_, Wb_hh_d, H_, gates, GH,
                                                    nullptr, nullptr, gates, GH, H_, 0);
        lstm_update<<<pw_grid, blk, 0, stream>>>(gates, hb, c);
        // z = tanh(h @ Wz^T + bz) -> out[:, t, :]
        wmma_gemm_async<64><<<gz, blk, 0, stream>>>(hb, H_, Wzb, H_, nullptr, 0,
                                                    bz, nullptr, out + (size_t)t * Z_,
                                                    T_ * Z_, H_, 1);
    }
}
